// FourLayer_64F_14826227106299
// MI455X (gfx1250) — matrix operand order repaired and checked by reading, not compiled, not run on hardware
//
#include <hip/hip_runtime.h>

typedef __attribute__((ext_vector_type(16))) _Float16 v16h;
typedef __attribute__((ext_vector_type(8)))  float    v8f;
typedef __attribute__((ext_vector_type(4)))  unsigned int v4u;
typedef __attribute__((ext_vector_type(4)))  _Float16 v4h;
typedef _Float16 half_t;

union AFrag { v16h h; v4u u4[2]; unsigned int w[8]; };
union HU { _Float16 h; unsigned short u; };

// ---------------- layer-1 input prep: NCHW f32 -> [n][82][82][4] f16, zero halo, c=3 zero --------
__global__ void k_prep(const float* __restrict__ in1, const float* __restrict__ in2,
                       half_t* __restrict__ dst) {
  long t = (long)blockIdx.x * blockDim.x + threadIdx.x;
  const long total = 1048L * 82 * 82;
  if (t >= total) return;
  int n  = (int)(t / 6724);
  int r  = (int)(t % 6724);
  int iy = r / 82, ix = r % 82;
  v4h o;
  o[0] = o[1] = o[2] = o[3] = (half_t)0.f;
  if (iy > 0 && iy < 81 && ix > 0 && ix < 81) {
    const float* src = (n < 1024) ? (in1 + (long)n * 3 * 6400)
                                  : (in2 + (long)(n - 1024) * 3 * 6400);
    int sp = (iy - 1) * 80 + (ix - 1);
    o[0] = (half_t)src[sp];
    o[1] = (half_t)src[6400 + sp];
    o[2] = (half_t)src[12800 + sp];
  }
  *(v4h*)(dst + t * 4) = o;
}

// ---------------- weight pre-pack into WMMA B-fragment order ------------------------------------
// B layout (16-bit B, 32x16): lane = N column (0-15), lanes 16-31 = K+8 half,
// dword v holds K pair ((v&3)*2 + (v>>2)*16 + {0,1}).  Global K ordering: mode1: K = tap*64 + c (C=64, Ktot=576)
//                                                      mode0: K = tap*4  + c (C=3 padded, Ktot=64)
__global__ void k_pack(const float* __restrict__ w, half_t* __restrict__ bp,
                       int ksteps, int mode) {
  int id  = blockIdx.x * blockDim.x + threadIdx.x;
  int per = ksteps * 256;
  if (id >= 4 * per) return;
  int t    = id / per;
  int rem  = id % per;
  int s    = rem >> 8;
  int lane = (rem >> 3) & 31;
  int v    = rem & 7;
  int o    = t * 16 + (lane & 15);
  int h    = lane >> 4;
  int Kb   = s * 32 + (v & 3) * 2 + (v >> 2) * 16 + h * 8;
  unsigned int pack = 0;
  for (int j = 0; j < 2; ++j) {
    int K = Kb + j;
    float val = 0.f;
    if (mode == 0) {
      int tap = K >> 2, c = K & 3;
      if (tap <= 8 && c < 3) val = w[(o * 3 + c) * 9 + tap];
    } else {
      int tap = K >> 6, c = K & 63;
      val = w[(o * 64 + c) * 9 + tap];
    }
    HU hu; hu.h = (_Float16)val;
    pack |= ((unsigned int)hu.u) << (16 * j);
  }
  ((unsigned int*)bp)[((long)(t * ksteps + s) * 32 + lane) * 8 + v] = pack;
}

// ---------------- implicit-GEMM conv via v_wmma_f32_16x16x32_f16 + fused BN partial sums --------
// One wave computes a 16(pixels) x 64(out-ch) tile.  A fragments loaded straight from the
// NHWC-halo activation buffer (2x b128 per K-step for C=64).  Output written as [pix][64] f16.
// Per-block per-channel (sum, sum^2) partials are LDS-reduced and written for the BN stats
// (deterministic: fixed block->pixel mapping, fixed reduce order).
// NOTE: grids must divide exactly (no partial blocks).
template<int CMODE, int KSTEPS>
__global__ __launch_bounds__(128)
void k_conv(const half_t* __restrict__ in, const half_t* __restrict__ bp,
            half_t* __restrict__ out, float* __restrict__ part,
            int outW, int outHW, int inW2, int inPlane) {
  __shared__ float red[2][4][32][4];     // [sum|sum2][wave][lane][ntile]
  int wv   = threadIdx.x >> 5;
  int wave = blockIdx.x * (blockDim.x >> 5) + wv;
  int lane = threadIdx.x & 31;
  int m0   = wave << 4;
  int r = lane & 15, h = lane >> 4;
  int pix = m0 + r;
  int n   = pix / outHW;
  int rem = pix - n * outHW;
  int y   = rem / outW;
  int x   = rem - y * outW;
  long pixBase = (long)n * inPlane + (long)(y * inW2 + x) * CMODE;

  v8f acc[4];
#pragma unroll
  for (int t = 0; t < 4; ++t)
#pragma unroll
    for (int g = 0; g < 8; ++g) acc[t][g] = 0.f;
  const long bstride = (long)KSTEPS * 512;   // elements per N-tile of packed B

  for (int s = 0; s < KSTEPS; ++s) {
    AFrag a;
    if (CMODE == 64) {
      int tap = s >> 1;
      int dy = tap / 3, dx = tap - dy * 3;
      int c0 = ((s & 1) << 5) + (h << 3);
      const half_t* p = in + pixBase + (long)(dy * inW2 + dx) * 64 + c0;
      a.u4[0] = *(const v4u*)p;
      a.u4[1] = *(const v4u*)(p + 16);
    } else {
#pragma unroll
      for (int v = 0; v < 8; ++v) {
        int K = s * 32 + ((v & 3) * 2) + ((v >> 2) * 16) + (h << 3);
        int tap = K >> 2; if (tap > 8) tap = 8;   // padded K region: B is zero there
        int c = K & 3;
        int dy = tap / 3, dx = tap - dy * 3;
        a.w[v] = *(const unsigned int*)(in + pixBase + (long)(dy * inW2 + dx) * 4 + c);
      }
    }
    const half_t* bs = bp + ((long)s * 32 + lane) * 16;
    AFrag b0, b1, b2, b3;
    b0.u4[0] = *(const v4u*)(bs);               b0.u4[1] = *(const v4u*)(bs + 8);
    b1.u4[0] = *(const v4u*)(bs + bstride);     b1.u4[1] = *(const v4u*)(bs + bstride + 8);
    b2.u4[0] = *(const v4u*)(bs + 2*bstride);   b2.u4[1] = *(const v4u*)(bs + 2*bstride + 8);
    b3.u4[0] = *(const v4u*)(bs + 3*bstride);   b3.u4[1] = *(const v4u*)(bs + 3*bstride + 8);
    acc[0] = __builtin_amdgcn_wmma_f32_16x16x32_f16(false, a.h, false, b0.h, (short)0, acc[0], false, false);
    acc[1] = __builtin_amdgcn_wmma_f32_16x16x32_f16(false, a.h, false, b1.h, (short)0, acc[1], false, false);
    acc[2] = __builtin_amdgcn_wmma_f32_16x16x32_f16(false, a.h, false, b2.h, (short)0, acc[2], false, false);
    acc[3] = __builtin_amdgcn_wmma_f32_16x16x32_f16(false, a.h, false, b3.h, (short)0, acc[3], false, false);
  }

  // store f16 conv output: D layout -> lanes 0-15 rows m0+g, lanes 16-31 rows m0+8+g, col lane&15
  int col = lane & 15;
  int rowOff = h << 3;
#pragma unroll
  for (int g = 0; g < 8; ++g) {
    long row = (long)m0 + g + rowOff;
    long ob  = row * 64 + col;
    out[ob +  0] = (half_t)acc[0][g];
    out[ob + 16] = (half_t)acc[1][g];
    out[ob + 32] = (half_t)acc[2][g];
    out[ob + 48] = (half_t)acc[3][g];
  }

  // fused BN partial sums (f32, pre-rounding)
#pragma unroll
  for (int t = 0; t < 4; ++t) {
    float s = 0.f, s2 = 0.f;
#pragma unroll
    for (int g = 0; g < 8; ++g) { float v = acc[t][g]; s += v; s2 += v * v; }
    red[0][wv][lane][t] = s;
    red[1][wv][lane][t] = s2;
  }
  __syncthreads();
  int nw = blockDim.x >> 5;
  if (threadIdx.x < 64) {
    int c = threadIdx.x;
    int t = c >> 4, cc = c & 15;
    float S = 0.f, S2 = 0.f;
    for (int w2 = 0; w2 < nw; ++w2) {
      S  += red[0][w2][cc][t] + red[0][w2][cc + 16][t];
      S2 += red[1][w2][cc][t] + red[1][w2][cc + 16][t];
    }
    part[((long)blockIdx.x * 64 + c) * 2]     = S;
    part[((long)blockIdx.x * 64 + c) * 2 + 1] = S2;
  }
}

// ---------------- stage-2 reduce of conv-block partials into 131 group-aligned chunks -----------
// chunk b covers conv blocks [b*chunkB, (b+1)*chunkB) = exactly 8 images.
// chunks 0..127 -> query group, 128..130 -> the three support sets.
__global__ void k_reduce(const float* __restrict__ part, float* __restrict__ part2, int chunkB) {
  __shared__ float sh[512];
  int b = blockIdx.x;                 // 131
  int tid = threadIdx.x;              // 256
  int c = tid & 63, q = tid >> 6;
  long b0 = (long)b * chunkB;
  float s = 0.f, s2 = 0.f;
  for (long k = b0 + q; k < b0 + chunkB; k += 4) {
    s  += part[k * 128 + c * 2];
    s2 += part[k * 128 + c * 2 + 1];
  }
  sh[tid] = s; sh[256 + tid] = s2;
  __syncthreads();
  if (q == 0) {
    float S  = sh[c] + sh[64 + c] + sh[128 + c] + sh[192 + c];
    float S2 = sh[256 + c] + sh[320 + c] + sh[384 + c] + sh[448 + c];
    part2[(b * 64 + c) * 2]     = S;
    part2[(b * 64 + c) * 2 + 1] = S2;
  }
}

__global__ void k_finalize(const float* __restrict__ part2, const float* __restrict__ gamma,
                           const float* __restrict__ beta, float* __restrict__ ss, int outHW) {
  int t = threadIdx.x;                // 256 = 4 groups x 64 channels
  int g = t >> 6, c = t & 63;
  int b0 = (g == 0) ? 0 : 127 + g;
  int nb = (g == 0) ? 128 : 1;
  float S = 0.f, S2 = 0.f;
  for (int i = 0; i < nb; ++i) {
    S  += part2[((b0 + i) * 64 + c) * 2];
    S2 += part2[((b0 + i) * 64 + c) * 2 + 1];
  }
  float cnt  = (float)((g == 0 ? 1024L : 8L) * outHW);
  float mean = S / cnt;
  float var  = S2 / cnt - mean * mean;
  float scale = gamma[c] * rsqrtf(var + 1e-5f);
  ss[(g * 64 + c) * 2]     = scale;
  ss[(g * 64 + c) * 2 + 1] = beta[c] - mean * scale;
}

// ---------------- zero the halo border of an NHWC-halo buffer (C=64) ----------------------------
__global__ void k_zero_halo(half_t* __restrict__ dst, int H2, int W2) {
  int perim = 2 * W2 + 2 * (H2 - 2);
  long t = (long)blockIdx.x * blockDim.x + threadIdx.x;
  long total = 1048L * perim;
  if (t >= total) return;
  int n = (int)(t / perim);
  int r = (int)(t % perim);
  int y, x;
  if (r < W2)            { y = 0;      x = r; }
  else if (r < 2 * W2)   { y = H2 - 1; x = r - W2; }
  else { int rr = r - 2 * W2; y = 1 + (rr >> 1); x = (rr & 1) ? (W2 - 1) : 0; }
  half_t* p = dst + ((long)(n * H2 + y) * W2 + x) * 64;
  v4u z = {0u, 0u, 0u, 0u};
#pragma unroll
  for (int i = 0; i < 8; ++i) *(v4u*)(p + i * 8) = z;
}

// ---------------- fused BN-apply + LeakyReLU(0.2) + 2x2 maxpool ---------------------------------
__global__ void k_pool(const half_t* __restrict__ conv, const float* __restrict__ ss,
                       half_t* __restrict__ dst, int outW, int outHW,
                       int poolW, int poolHW, int halo /* = W2, or 0 for no-halo output */) {
  long t = (long)blockIdx.x * blockDim.x + threadIdx.x;
  long total = 1048L * poolHW * 64;
  if (t >= total) return;
  int  c  = (int)(t & 63);
  long p  = t >> 6;
  int  n  = (int)(p / poolHW);
  int  rr = (int)(p - (long)n * poolHW);
  int  py = rr / poolW, px = rr - py * poolW;
  int  g  = (n < 1024) ? 0 : 1 + ((n - 1024) >> 3);
  float sc = ss[(g * 64 + c) * 2], sf = ss[(g * 64 + c) * 2 + 1];
  long base = ((long)n * outHW + (long)(2 * py) * outW + 2 * px) * 64 + c;
  float m = -3.4e38f;
#pragma unroll
  for (int j = 0; j < 2; ++j)
#pragma unroll
    for (int i = 0; i < 2; ++i) {
      float v = (float)conv[base + (long)(j * outW + i) * 64];
      v = v * sc + sf;
      v = (v > 0.f) ? v : 0.2f * v;
      m = (v > m) ? v : m;
    }
  long o;
  if (halo) {
    int W2 = halo;
    int poolH = poolHW / poolW;
    o = ((long)(n * (poolH + 2) + py + 1) * W2 + px + 1) * 64 + c;
  } else {
    o = t;
  }
  dst[o] = (half_t)m;
}

// ---------------- 5x5 maxpool -> 64-d embedding (f32) -------------------------------------------
__global__ void k_embed(const half_t* __restrict__ feat, float* __restrict__ emb) {
  int t = blockIdx.x * blockDim.x + threadIdx.x;
  if (t >= 1048 * 64) return;
  int n = t >> 6, c = t & 63;
  const half_t* p = feat + (long)n * 25 * 64 + c;
  float m = -3.4e38f;
  for (int i = 0; i < 25; ++i) { float v = (float)p[i * 64]; m = (v > m) ? v : m; }
  emb[t] = m;
}

// ---------------- prototypes + self-attention + LayerNorm + row-normalize (3x64, one block) -----
__global__ void k_attn(const float* __restrict__ emb,
                       const float* __restrict__ wq, const float* __restrict__ wk,
                       const float* __restrict__ wv, const float* __restrict__ fcw,
                       const float* __restrict__ fcb, const float* __restrict__ lng,
                       const float* __restrict__ lnb, float* __restrict__ sn) {
  __shared__ float S[3][64], QP[3][64], KP[3][64], VP[3][64], OV[3][64], RS[3][64];
  __shared__ float attn[9], mu[3], rstd[3], rnorm[3];
  int e = threadIdx.x;  // 64 threads
  for (int l = 0; l < 3; ++l) {
    float s = 0.f;
    for (int j = 0; j < 8; ++j) s += emb[(1024 + l * 8 + j) * 64 + e];
    S[l][e] = s * 0.125f;
  }
  __syncthreads();
  for (int l = 0; l < 3; ++l) {
    float q = 0.f, k = 0.f, v = 0.f;
    for (int d = 0; d < 64; ++d) {
      float sv = S[l][d];
      q += sv * wq[e * 64 + d];
      k += sv * wk[e * 64 + d];
      v += sv * wv[e * 64 + d];
    }
    QP[l][e] = q; KP[l][e] = k; VP[l][e] = v;
  }
  __syncthreads();
  if (e < 9) {
    int qi = e / 3, ki = e % 3;
    float a = 0.f;
    for (int d = 0; d < 64; ++d) a += QP[qi][d] * KP[ki][d];
    attn[e] = a * 0.125f;   // / sqrt(64)
  }
  __syncthreads();
  if (e < 3) {
    float a0 = attn[e*3], a1 = attn[e*3+1], a2 = attn[e*3+2];
    float mx = fmaxf(a0, fmaxf(a1, a2));
    float e0 = expf(a0 - mx), e1 = expf(a1 - mx), e2 = expf(a2 - mx);
    float inv = 1.f / (e0 + e1 + e2);
    attn[e*3] = e0 * inv; attn[e*3+1] = e1 * inv; attn[e*3+2] = e2 * inv;
  }
  __syncthreads();
  for (int l = 0; l < 3; ++l)
    OV[l][e] = attn[l*3] * VP[0][e] + attn[l*3+1] * VP[1][e] + attn[l*3+2] * VP[2][e];
  __syncthreads();
  for (int l = 0; l < 3; ++l) {
    float a = fcb[e];
    for (int d = 0; d < 64; ++d) a += OV[l][d] * fcw[e * 64 + d];
    RS[l][e] = a + S[l][e];
  }
  __syncthreads();
  if (e < 3) {
    float m = 0.f;
    for (int d = 0; d < 64; ++d) m += RS[e][d];
    m *= (1.f / 64.f);
    float v = 0.f;
    for (int d = 0; d < 64; ++d) { float dd = RS[e][d] - m; v += dd * dd; }
    v *= (1.f / 64.f);
    mu[e] = m; rstd[e] = rsqrtf(v + 1e-5f);
  }
  __syncthreads();
  for (int l = 0; l < 3; ++l)
    RS[l][e] = (RS[l][e] - mu[l]) * rstd[l] * lng[e] + lnb[e];
  __syncthreads();
  if (e < 3) {
    float s = 0.f;
    for (int d = 0; d < 64; ++d) s += RS[e][d] * RS[e][d];
    rnorm[e] = rsqrtf(s);
  }
  __syncthreads();
  for (int l = 0; l < 3; ++l) sn[l * 64 + e] = RS[l][e] * rnorm[l];
}

// ---------------- cosine similarity: out[n][s] = (q_n . sn_s) / |q_n| ---------------------------
__global__ void k_sim(const float* __restrict__ emb, const float* __restrict__ sn,
                      float* __restrict__ out) {
  int n = blockIdx.x * blockDim.x + threadIdx.x;
  if (n >= 1024) return;
  float s0 = 0.f, s1 = 0.f, s2 = 0.f, nn = 0.f;
  for (int d = 0; d < 64; ++d) {
    float q = emb[n * 64 + d];
    nn += q * q;
    s0 += q * sn[d]; s1 += q * sn[64 + d]; s2 += q * sn[128 + d];
  }
  float inv = rsqrtf(nn);
  out[n * 3]     = s0 * inv;
  out[n * 3 + 1] = s1 * inv;
  out[n * 3 + 2] = s2 * inv;
}

// ================================================================================================
extern "C" void kernel_launch(void* const* d_in, const int* in_sizes, int n_in,
                              void* d_out, int out_size, void* d_ws, size_t ws_size,
                              hipStream_t stream) {
  (void)in_sizes; (void)n_in; (void)out_size; (void)ws_size;
  const float* in1 = (const float*)d_in[0];
  const float* in2 = (const float*)d_in[1];
  const float* w[4]   = {(const float*)d_in[2], (const float*)d_in[3],
                         (const float*)d_in[4], (const float*)d_in[5]};
  const float* bng[4] = {(const float*)d_in[6], (const float*)d_in[8],
                         (const float*)d_in[10], (const float*)d_in[12]};
  const float* bnb[4] = {(const float*)d_in[7], (const float*)d_in[9],
                         (const float*)d_in[11], (const float*)d_in[13]};
  const float* wq  = (const float*)d_in[14];
  const float* wk  = (const float*)d_in[15];
  const float* wv  = (const float*)d_in[16];
  const float* fcw = (const float*)d_in[17];
  const float* fcb = (const float*)d_in[18];
  const float* lng = (const float*)d_in[19];
  const float* lnb = (const float*)d_in[20];

  char* ws = (char*)d_ws;
  float*  EMB   = (float*)ws;                        // 1048*64*4       = 268288
  float*  SN    = (float*)(ws + 268288);             // 3*64*4          = 768
  float*  PART2 = (float*)(ws + 269056);             // 131*64*2*4      = 67072
  float*  SS    = (float*)(ws + 336128);             // 4*64*2*4        = 2048
  half_t* BP1   = (half_t*)(ws + 338176);            // 4*2*32*32       = 8192
  half_t* BP2   = (half_t*)(ws + 346368);            // 4*18*32*32      = 73728
  half_t* BP3   = (half_t*)(ws + 420096);
  half_t* BP4   = (half_t*)(ws + 493824);
  float*  PART  = (float*)(ws + (1L << 20));                             // <= 53.7 MB
  half_t* INA   = (half_t*)(ws + (56L << 20));                           // <= 62 MB
  half_t* INB   = (half_t*)(ws + (128L << 20));                          // <= 226 MB
  half_t* CONV  = (half_t*)(ws + (368L << 20));                          // <= 820 MB

  // layer-1 input (NHWC4, halo) + weight packs
  { long tot = 1048L * 82 * 82;
    k_prep<<<(unsigned)((tot + 255) / 256), 256, 0, stream>>>(in1, in2, INA); }
  k_pack<<<(4 * 2  * 256) / 256, 256, 0, stream>>>(w[0], BP1, 2,  0);
  k_pack<<<(4 * 18 * 256) / 256, 256, 0, stream>>>(w[1], BP2, 18, 1);
  k_pack<<<(4 * 18 * 256) / 256, 256, 0, stream>>>(w[2], BP3, 18, 1);
  k_pack<<<(4 * 18 * 256) / 256, 256, 0, stream>>>(w[3], BP4, 18, 1);

  // ---- layer 1: conv 80x80 (+stats) -> pool -> [1048,42,42,64]h ----
  { // 1048*6400 px / 64 px-per-block = 104800 blocks (exact)
    k_conv<4, 2><<<104800, 128, 0, stream>>>(INA, BP1, CONV, PART, 80, 6400, 82, 26896);
    k_reduce<<<131, 256, 0, stream>>>(PART, PART2, 800);
    k_finalize<<<1, 256, 0, stream>>>(PART2, bng[0], bnb[0], SS, 6400);
    { long tot = 1048L * (2 * 42 + 2 * 40);
      k_zero_halo<<<(unsigned)((tot + 255) / 256), 256, 0, stream>>>(INB, 42, 42); }
    { long tot = 1048L * 1600 * 64;
      k_pool<<<(unsigned)((tot + 255) / 256), 256, 0, stream>>>(
          CONV, SS, INB, 80, 6400, 40, 1600, 42); } }

  // ---- layer 2: 40x40 -> pool -> [1048,22,22,64]h ----
  { // 1048*1600/64 = 26200 blocks (exact)
    k_conv<64, 18><<<26200, 128, 0, stream>>>(INB, BP2, CONV, PART, 40, 1600, 42, 112896);
    k_reduce<<<131, 256, 0, stream>>>(PART, PART2, 200);
    k_finalize<<<1, 256, 0, stream>>>(PART2, bng[1], bnb[1], SS, 1600);
    { long tot = 1048L * (2 * 22 + 2 * 20);
      k_zero_halo<<<(unsigned)((tot + 255) / 256), 256, 0, stream>>>(INA, 22, 22); }
    { long tot = 1048L * 400 * 64;
      k_pool<<<(unsigned)((tot + 255) / 256), 256, 0, stream>>>(
          CONV, SS, INA, 40, 1600, 20, 400, 22); } }

  // ---- layer 3: 20x20 -> pool -> [1048,12,12,64]h ----
  { // 1048*400/64 = 6550 blocks (exact)
    k_conv<64, 18><<<6550, 128, 0, stream>>>(INA, BP3, CONV, PART, 20, 400, 22, 30976);
    k_reduce<<<131, 256, 0, stream>>>(PART, PART2, 50);
    k_finalize<<<1, 256, 0, stream>>>(PART2, bng[2], bnb[2], SS, 400);
    { long tot = 1048L * (2 * 12 + 2 * 10);
      k_zero_halo<<<(unsigned)((tot + 255) / 256), 256, 0, stream>>>(INB, 12, 12); }
    { long tot = 1048L * 100 * 64;
      k_pool<<<(unsigned)((tot + 255) / 256), 256, 0, stream>>>(
          CONV, SS, INB, 20, 400, 10, 100, 12); } }

  // ---- layer 4: 10x10 (2 waves/block so 8-image groups stay block-aligned) -> [1048,25,64]h ----
  { // 1048*100/32 = 3275 blocks (exact); 8 images = 25 blocks
    k_conv<64, 18><<<3275, 64, 0, stream>>>(INB, BP4, CONV, PART, 10, 100, 12, 9216);
    k_reduce<<<131, 256, 0, stream>>>(PART, PART2, 25);
    k_finalize<<<1, 256, 0, stream>>>(PART2, bng[3], bnb[3], SS, 100);
    { long tot = 1048L * 25 * 64;
      k_pool<<<(unsigned)((tot + 255) / 256), 256, 0, stream>>>(
          CONV, SS, INA, 10, 100, 5, 25, 0); } }

  // ---- head ----
  k_embed<<<(1048 * 64 + 255) / 256, 256, 0, stream>>>(INA, EMB);
  k_attn<<<1, 64, 0, stream>>>(EMB, wq, wk, wv, fcw, fcb, lng, lnb, SN);
  k_sim<<<4, 256, 0, stream>>>(EMB, SN, (float*)d_out);
}
